// Net_90967407329979
// MI455X (gfx1250) — compile-verified, matrix operand order checked by reading, not run on hardware
//
#include <hip/hip_runtime.h>
#include <math.h>

// ---------------- types ----------------
typedef __attribute__((ext_vector_type(16))) __bf16 v16bf;
typedef __attribute__((ext_vector_type(8)))  float  v8f;
typedef __attribute__((ext_vector_type(4)))  unsigned int v4u;
typedef __attribute__((ext_vector_type(8)))  int v8i;
typedef __attribute__((ext_vector_type(4)))  int v4i;

union FragB16 {
    v16bf              v;
    unsigned long long q[4];
};

__device__ __forceinline__ unsigned short f2bf(float f) {   // RNE fp32->bf16
    unsigned int u = __float_as_uint(f);
    unsigned int r = u + 0x7fffu + ((u >> 16) & 1u);
    return (unsigned short)(r >> 16);
}
__device__ __forceinline__ unsigned lds_off(const void* p) { // generic->LDS byte offset
    return (unsigned)(unsigned long long)p;                  // low 32 bits of the generic address hold the LDS offset
}

// ---------------- constants ----------------
#define BATCH   65536
#define HID     246
#define LDH     256       // padded activation stride (fp32 row 1024B / bf16 row 512B)
#define LDW1    800       // padded bf16 W1 stride (784 -> 800, mult of 32, 16B rows)
#define TM      128
#define TN2     256
#define TK      32
#define LDSA_ST 40        // 32 data + 4DW pad -> 80B stride, conflict-free, 16B aligned
#define LDSB_ST 40
#define ALPHA_L 0.1f

// ---------------- Tensor Data Mover ----------------
#if defined(__has_builtin)
#  if __has_builtin(__builtin_amdgcn_tensor_load_to_lds)
#    define HAVE_TDM 1
#  endif
#endif
#ifndef HAVE_TDM
#  define HAVE_TDM 0
#endif

#if HAVE_TDM
#if __has_include(<hip/amd_detail/amd_gfx1250_TDM.h>)
#  define TDM_CALL(g0, g1) __builtin_amdgcn_tensor_load_to_lds(g0, g1, (v4i)0, (v4i)0, (v8i)0, 0)
#else
#  define TDM_CALL(g0, g1) __builtin_amdgcn_tensor_load_to_lds(g0, g1, (v4i)0, (v4i)0, 0)
#endif

// 2D bf16 tile load: tileRows x 32 elems, row stride 'strideElems', LDS gets
// 64B data + 16B pad per row (pad_interval=16DW, pad_amount=4DW) == 40-elem rows.
__device__ __forceinline__ void tdm_load_2d(unsigned ldsAddr,
                                            const unsigned short* gptr,
                                            int tileRows, int strideElems)
{
    const unsigned long long g = (unsigned long long)(const void*)gptr;
    v4u g0;
    g0[0] = (unsigned)__builtin_amdgcn_readfirstlane(1);                 // count=1
    g0[1] = (unsigned)__builtin_amdgcn_readfirstlane((int)ldsAddr);      // lds_addr
    g0[2] = (unsigned)__builtin_amdgcn_readfirstlane((int)(unsigned)g);  // gaddr lo
    g0[3] = (unsigned)__builtin_amdgcn_readfirstlane(
                (int)(((unsigned)(g >> 32)) | 0x80000000u));             // gaddr hi | type=2
    v8i g1;
    g1[0] = __builtin_amdgcn_readfirstlane(
                (int)((1u << 16) | (1u << 20) | (3u << 22) | (3u << 25))); // 2B, pad 4DW/16DW
    g1[1] = __builtin_amdgcn_readfirstlane((int)(TK << 16));             // tensor_dim0=32
    g1[2] = __builtin_amdgcn_readfirstlane((int)((tileRows & 0xffff) << 16)); // tensor_dim1
    g1[3] = __builtin_amdgcn_readfirstlane((int)(TK << 16));             // tile_dim0=32
    g1[4] = __builtin_amdgcn_readfirstlane(tileRows & 0xffff);           // tile_dim1
    g1[5] = __builtin_amdgcn_readfirstlane(strideElems);                 // dim0_stride
    g1[6] = 0;
    g1[7] = 0;
    TDM_CALL(g0, g1);
}
#endif // HAVE_TDM

// ---------------- shared GEMM pieces ----------------
// wave tile 32(M) x 128(N): 2x8 accumulators; block = 8 waves = 128x256
__device__ __forceinline__ void mma_tile(const unsigned short* bufA,
                                         const unsigned short* bufB,
                                         int waveM, int waveN, int lane,
                                         v8f (&acc)[2][8])
{
    const int kb = (lane >> 4) * 8;
    const int rl = lane & 15;
    FragB16 a[2];
    #pragma unroll
    for (int mi = 0; mi < 2; ++mi) {
        const unsigned short* p0 = bufA + (waveM * 32 + mi * 16 + rl) * LDSA_ST + kb;
        a[mi].q[0] = *(const unsigned long long*)(p0);
        a[mi].q[1] = *(const unsigned long long*)(p0 + 4);
        a[mi].q[2] = *(const unsigned long long*)(p0 + 16);
        a[mi].q[3] = *(const unsigned long long*)(p0 + 20);
    }
    #pragma unroll
    for (int ni = 0; ni < 8; ++ni) {
        const unsigned short* p0 = bufB + (waveN * 128 + ni * 16 + rl) * LDSB_ST + kb;
        FragB16 b;
        b.q[0] = *(const unsigned long long*)(p0);
        b.q[1] = *(const unsigned long long*)(p0 + 4);
        b.q[2] = *(const unsigned long long*)(p0 + 16);
        b.q[3] = *(const unsigned long long*)(p0 + 20);
        #pragma unroll
        for (int mi = 0; mi < 2; ++mi)
            acc[mi][ni] = __builtin_amdgcn_wmma_f32_16x16x32_bf16(
                false, a[mi].v, false, b.v, (short)0, acc[mi][ni], false, false);
    }
}

__device__ __forceinline__ void epilogue(v8f (&acc)[2][8], int waveM, int waveN,
                                         int lane, int mBlock, const float* bias,
                                         float* outF, unsigned short* outB,
                                         int ldo, int N)
{
    const int rl  = lane & 15;
    const int rhi = (lane >> 4) * 8;
    #pragma unroll
    for (int ni = 0; ni < 8; ++ni) {
        const int  col = waveN * 128 + ni * 16 + rl;
        const bool ok  = col < N;
        const float bv = ok ? bias[col] : 0.f;
        #pragma unroll
        for (int mi = 0; mi < 2; ++mi) {
            const int row0 = mBlock + waveM * 32 + mi * 16 + rhi;
            #pragma unroll
            for (int r = 0; r < 8; ++r) {
                float v = acc[mi][ni][r] + bv;
                v = fmaxf(v, ALPHA_L * v);                 // leaky_relu
                const size_t off = (size_t)(row0 + r) * (size_t)ldo + col;
                if (ok) outF[off] = v;
                outB[off] = ok ? f2bf(v) : (unsigned short)0;  // zero-padded bf16 copy
            }
        }
    }
}

// =====================================================================
// Layer 1: A = x (fp32, converted in-kernel), W = pre-converted bf16 (TDM)
// =====================================================================
__global__ __launch_bounds__(256)
void gemm_l1(const float* __restrict__ A, int lda,
             const unsigned short* __restrict__ Wb, int ldw,
             const float* __restrict__ bias,
             float* __restrict__ outF, unsigned short* __restrict__ outB,
             int ldo, int N, int K)
{
    __shared__ unsigned short lsA[TM * LDSA_ST];
    __shared__ unsigned short lsB[TN2 * LDSB_ST];

    const int tid = threadIdx.x, lane = tid & 31, wave = tid >> 5;
    const int waveM = wave & 3, waveN = wave >> 2;
    const int mBlock = blockIdx.x * TM;

    v8f acc[2][8] = {};
    const int kq = tid & 7, rr = tid >> 3;
    const int nChunks = (K + TK - 1) / TK;       // 25

    for (int c = 0; c < nChunks; ++c) {
        const int k0 = c * TK;
        __syncthreads();                          // prior compute done
#if HAVE_TDM
        if (wave == 0) tdm_load_2d(lds_off(&lsB[0]), Wb + k0, TN2, ldw);
#else
        #pragma unroll
        for (int i = 0; i < 4; ++i) {             // 256x32 bf16 copy
            const int q = tid + 256 * i, row = q >> 2, pc = q & 3;
            *(uint4*)&lsB[row * LDSB_ST + pc * 8] =
                *(const uint4*)(Wb + (size_t)row * ldw + k0 + pc * 8);
        }
#endif
        // manual A stage: 128x32 fp32 -> bf16 (K=784 tail checked)
        #pragma unroll
        for (int p = 0; p < 4; ++p) {
            const int r  = rr + 32 * p;
            const int kk = k0 + kq * 4;
            const float* src = A + (size_t)(mBlock + r) * (size_t)lda + kk;
            float4 v;
            if (kk + 3 < K) v = *(const float4*)src;
            else {
                v.x = (kk + 0 < K) ? src[0] : 0.f;
                v.y = (kk + 1 < K) ? src[1] : 0.f;
                v.z = (kk + 2 < K) ? src[2] : 0.f;
                v.w = (kk + 3 < K) ? src[3] : 0.f;
            }
            unsigned short* d = &lsA[r * LDSA_ST + kq * 4];
            d[0] = f2bf(v.x); d[1] = f2bf(v.y); d[2] = f2bf(v.z); d[3] = f2bf(v.w);
        }
#if HAVE_TDM
        if (wave == 0) __builtin_amdgcn_s_wait_tensorcnt(0);
#endif
        __syncthreads();
        mma_tile(lsA, lsB, waveM, waveN, lane, acc);
    }
    epilogue(acc, waveM, waveN, lane, mBlock, bias, outF, outB, ldo, N);
}

// =====================================================================
// Layers 2-6: A and W both bf16 -> double-buffered TDM staging
// =====================================================================
__global__ __launch_bounds__(256)
void gemm_bf16(const unsigned short* __restrict__ Abf, int lda,
               const unsigned short* __restrict__ Wb, int ldw,
               const float* __restrict__ bias,
               float* __restrict__ outF, unsigned short* __restrict__ outB,
               int ldo, int N, int K)
{
    __shared__ unsigned short lsA[2 * TM * LDSA_ST];
    __shared__ unsigned short lsB[2 * TN2 * LDSB_ST];

    const int tid = threadIdx.x, lane = tid & 31, wave = tid >> 5;
    const int waveM = wave & 3, waveN = wave >> 2;
    const int mBlock = blockIdx.x * TM;

    v8f acc[2][8] = {};
    const int nChunks = (K + TK - 1) / TK;       // 8 (K padded data exists to 256)

#if HAVE_TDM
    if (wave == 0) {                              // prologue: chunk 0 -> buffer 0
        tdm_load_2d(lds_off(&lsA[0]), Abf + (size_t)mBlock * lda, TM, lda);
        tdm_load_2d(lds_off(&lsB[0]), Wb, TN2, ldw);
    }
    for (int c = 0; c < nChunks; ++c) {
        if (wave == 0) __builtin_amdgcn_s_wait_tensorcnt(0);
        __syncthreads();                          // chunk c visible; prev compute done
        if (wave == 0 && c + 1 < nChunks) {       // prefetch chunk c+1 (other buffer)
            const int k0n = (c + 1) * TK;
            const int nb  = (c + 1) & 1;
            tdm_load_2d(lds_off(&lsA[nb * TM * LDSA_ST]),
                        Abf + (size_t)mBlock * lda + k0n, TM, lda);
            tdm_load_2d(lds_off(&lsB[nb * TN2 * LDSB_ST]), Wb + k0n, TN2, ldw);
        }
        mma_tile(&lsA[(c & 1) * TM * LDSA_ST], &lsB[(c & 1) * TN2 * LDSB_ST],
                 waveM, waveN, lane, acc);
    }
#else
    for (int c = 0; c < nChunks; ++c) {
        const int k0 = c * TK;
        __syncthreads();
        #pragma unroll
        for (int i = 0; i < 2; ++i) {             // A: 128x32 bf16
            const int q = tid + 256 * i, row = q >> 2, pc = q & 3;
            *(uint4*)&lsA[row * LDSA_ST + pc * 8] =
                *(const uint4*)(Abf + (size_t)(mBlock + row) * lda + k0 + pc * 8);
        }
        #pragma unroll
        for (int i = 0; i < 4; ++i) {             // B: 256x32 bf16
            const int q = tid + 256 * i, row = q >> 2, pc = q & 3;
            *(uint4*)&lsB[row * LDSB_ST + pc * 8] =
                *(const uint4*)(Wb + (size_t)row * ldw + k0 + pc * 8);
        }
        __syncthreads();
        mma_tile(lsA, lsB, waveM, waveN, lane, acc);
    }
#endif
    epilogue(acc, waveM, waveN, lane, mBlock, bias, outF, outB, ldo, N);
}

// =====================================================================
// Weight pre-convert: fp32 [N x K] -> bf16 [rows x ldw], zero padded
// =====================================================================
__global__ __launch_bounds__(256)
void convert_w(const float* __restrict__ W, int N, int K,
               unsigned short* __restrict__ Wb, int ldw, int total)
{
    for (int i = blockIdx.x * 256 + threadIdx.x; i < total; i += gridDim.x * 256) {
        const int r = i / ldw, k = i - r * ldw;
        const float v = (r < N && k < K) ? W[(size_t)r * K + k] : 0.f;
        Wb[i] = f2bf(v);
    }
}

// =====================================================================
// Entropy: one wave per row; LDS histogram; deterministic 2-stage mean
// =====================================================================
__global__ __launch_bounds__(256)
void entropy_partial(const float* __restrict__ h, int ldh, int D,
                     float* __restrict__ partials)
{
    __shared__ unsigned int cnt[8][100];
    __shared__ float        wsum[8];

    const int wave = threadIdx.x >> 5, lane = threadIdx.x & 31;
    const size_t row = (size_t)blockIdx.x * 8 + wave;
    const float* hr = h + row * (size_t)ldh;

    for (int j = lane; j < 100; j += 32) cnt[wave][j] = 0u;

    float mn = 3.4e38f, mx = -3.4e38f;
    for (int k = lane; k < D; k += 32) {
        float v = hr[k];
        mn = fminf(mn, v); mx = fmaxf(mx, v);
    }
    #pragma unroll
    for (int off = 16; off >= 1; off >>= 1) {
        mn = fminf(mn, __shfl_xor(mn, off, 32));
        mx = fmaxf(mx, __shfl_xor(mx, off, 32));
    }
    const float range = mx - mn;
    const float rcp   = (range > 0.f) ? (1.0f / range) : 0.0f;

    for (int k = lane; k < D; k += 32) {
        float hn = (hr[k] - mn) * rcp;
        int idx = (int)floorf(hn * 100.0f);
        idx = idx < 0 ? 0 : (idx > 99 ? 99 : idx);
        atomicAdd(&cnt[wave][idx], 1u);
    }
    __syncthreads();

    const float invD = 1.0f / (float)D;
    float s = 0.f;
    for (int j = lane; j < 100; j += 32) {
        float c = (float)cnt[wave][j];
        if (c > 0.f) { float p = c * invD; s += p * __logf(p); }
    }
    #pragma unroll
    for (int off = 16; off >= 1; off >>= 1) s += __shfl_xor(s, off, 32);
    if (lane == 0) wsum[wave] = -s + __logf(100.0f);
    __syncthreads();
    if (threadIdx.x == 0) {
        float t = 0.f;
        #pragma unroll
        for (int w = 0; w < 8; ++w) t += wsum[w];
        partials[blockIdx.x] = t;
    }
}

__global__ __launch_bounds__(256)
void reduce_mean(const float* __restrict__ partials, int n, float invB,
                 float* __restrict__ dst)
{
    __shared__ float sh[256];
    float s = 0.f;
    for (int i = threadIdx.x; i < n; i += 256) s += partials[i];
    sh[threadIdx.x] = s; __syncthreads();
    for (int st = 128; st > 0; st >>= 1) {
        if (threadIdx.x < st) sh[threadIdx.x] += sh[threadIdx.x + st];
        __syncthreads();
    }
    if (threadIdx.x == 0) *dst = sh[0] * invB;
}

// =====================================================================
// Final: relu(h @ W7^T + b7) -> log_softmax, one wave per row
// =====================================================================
__global__ __launch_bounds__(256)
void final_layer(const float* __restrict__ h, int ldh,
                 const float* __restrict__ W7, const float* __restrict__ b7,
                 float* __restrict__ out)
{
    const int wave = threadIdx.x >> 5, lane = threadIdx.x & 31;
    const size_t row = (size_t)blockIdx.x * 8 + wave;
    const float* hr = h + row * (size_t)ldh;

    float acc[10];
    #pragma unroll
    for (int n = 0; n < 10; ++n) acc[n] = 0.f;
    for (int k = lane; k < HID; k += 32) {
        const float hv = hr[k];
        #pragma unroll
        for (int n = 0; n < 10; ++n) acc[n] += hv * W7[n * HID + k];
    }
    #pragma unroll
    for (int n = 0; n < 10; ++n)
        #pragma unroll
        for (int off = 16; off >= 1; off >>= 1)
            acc[n] += __shfl_xor(acc[n], off, 32);

    if (lane == 0) {
        float l[10], m = -3.4e38f;
        #pragma unroll
        for (int n = 0; n < 10; ++n) {
            l[n] = fmaxf(acc[n] + b7[n], 0.f);
            m = fmaxf(m, l[n]);
        }
        float se = 0.f;
        #pragma unroll
        for (int n = 0; n < 10; ++n) se += __expf(l[n] - m);
        const float lse = m + __logf(se);
        #pragma unroll
        for (int n = 0; n < 10; ++n) out[row * 10 + n] = l[n] - lse;
    }
}

// =====================================================================
// launch
// =====================================================================
extern "C" void kernel_launch(void* const* d_in, const int* in_sizes, int n_in,
                              void* d_out, int out_size, void* d_ws, size_t ws_size,
                              hipStream_t stream)
{
    (void)in_sizes; (void)n_in; (void)ws_size;

    const float* x = (const float*)d_in[0];
    const float* Wl[7]; const float* bl[7];
    for (int i = 0; i < 7; ++i) {
        Wl[i] = (const float*)d_in[1 + 2 * i];
        bl[i] = (const float*)d_in[2 + 2 * i];
    }

    // workspace layout
    float*          hF  = (float*)d_ws;                              // 65536x256 f32
    unsigned short* hbA = (unsigned short*)(hF + (size_t)BATCH * LDH);
    unsigned short* hbB = hbA + (size_t)BATCH * LDH;                 // bf16 ping/pong
    unsigned short* w1b = hbB + (size_t)BATCH * LDH;                 // 256 x 800
    unsigned short* wkb = w1b + (size_t)256 * LDW1;                  // 5 x 256 x 256
    float* partials     = (float*)(wkb + (size_t)5 * 256 * LDH);

    float* out = (float*)d_out;
    float* ent = out + (out_size - 6);

    const dim3 blk(256);
    const int  nEnt = BATCH / 8;                 // 8192 blocks / partials

    // one-time weight conversion (bf16, zero-padded to 256 rows / padded K)
    convert_w<<<dim3(800), blk, 0, stream>>>(Wl[0], HID, 784, w1b, LDW1, 256 * LDW1);
    for (int l = 1; l < 6; ++l)
        convert_w<<<dim3(256), blk, 0, stream>>>(Wl[l], HID, HID,
                                                 wkb + (size_t)(l - 1) * 256 * LDH,
                                                 LDH, 256 * LDH);

    // layer 1
    gemm_l1<<<dim3(BATCH / TM), blk, 0, stream>>>(x, 784, w1b, LDW1, bl[0],
                                                  hF, hbA, LDH, HID, 784);
    entropy_partial<<<dim3(nEnt), blk, 0, stream>>>(hF, LDH, HID, partials);
    reduce_mean<<<dim3(1), blk, 0, stream>>>(partials, nEnt, 1.0f / BATCH, ent + 0);

    // layers 2-6
    const unsigned short* cur = hbA;
    for (int l = 1; l < 6; ++l) {
        unsigned short* nxt = (l & 1) ? hbB : hbA;
        gemm_bf16<<<dim3(BATCH / TM), blk, 0, stream>>>(cur, LDH,
                                                        wkb + (size_t)(l - 1) * 256 * LDH, LDH,
                                                        bl[l], hF, nxt, LDH, HID, HID);
        entropy_partial<<<dim3(nEnt), blk, 0, stream>>>(hF, LDH, HID, partials);
        reduce_mean<<<dim3(1), blk, 0, stream>>>(partials, nEnt, 1.0f / BATCH, ent + l);
        cur = nxt;
    }

    final_layer<<<dim3(BATCH / 8), blk, 0, stream>>>(hF, LDH, Wl[6], bl[6], out);
}
